// MyRNN_23398981829095
// MI455X (gfx1250) — compile-verified, matrix operand order checked by reading, not run on hardware
//
#include <hip/hip_runtime.h>

// ---------------------------------------------------------------------------
// 2-layer LSTM (Keras gate order i,f,g,o) on MI455X / gfx1250.
//
// Roofline: bf16-packed weights = 26 MB -> resident in the 192 MB L2 across
// all 160 recurrent steps; HBM traffic negligible. The chip is limited by
// WMMA issue + memory latency on the serial recurrent chain, so:
//  * bf16 WMMA (v_wmma_f32_16x16x32_bf16), f32 accumulate.
//  * B (weight) tiles staged into LDS with CDNA5 async global->LDS DMA
//    (global_load_async_to_lds_b128 / ASYNCcnt), double-buffered: DMA of
//    chunk c+1 overlaps the 8 WMMAs of chunk c; 4 waves share each tile.
//  * All 8 B fragments + 2 A fragments of a chunk are loaded into distinct
//    registers before the WMMA burst so ds/global latency hides behind
//    partial s_wait counts instead of wait-0 stalls.
//  * Last chunk peeled so the staging DMA is unconditional in the loops
//    (no per-iteration branch diamond).
//  * Gate activations + cell update fused in the GEMM epilogue; h state
//    ping-pongs per step, c updated in place (block-private elements).
// ---------------------------------------------------------------------------

typedef __attribute__((ext_vector_type(16))) __bf16    v16bf;
typedef __attribute__((ext_vector_type(8)))  float     v8f;
typedef __attribute__((ext_vector_type(4)))  unsigned  v4u;

union Frag {
    v16bf bf;
    v4u   q[2];
};

// Problem constants
#define BATCH  512
#define TSTEPS 80
#define EMBD   100
#define EMBP   128     // EMB padded to K multiple of 64
#define UNITS  1024
#define NOUT   4096    // 4*UNITS
#define KTOT1  (EMBP + UNITS)   // 1152
#define KTOT2  (UNITS + UNITS)  // 2048

// LDS staging geometry
#define KC        64            // K elements staged per barrier (2 WMMA k-steps)
#define BROWS     128           // 4 gates * 32 units per block
#define BSTRIDE   72            // padded row stride (elems): bank starts 36*r%64
                                // all distinct, spacing 4 -> conflict-free b128
#define BUF_ELEMS (BROWS * BSTRIDE)

__device__ __forceinline__ float fsig(float x) {
    return 1.0f / (1.0f + __expf(-x));
}
__device__ __forceinline__ float ftanh(float x) {
    x = fminf(fmaxf(x, -15.0f), 15.0f);
    float e = __expf(2.0f * x);
    return (e - 1.0f) / (e + 1.0f);
}

__device__ __forceinline__ v8f wmma_bf16(const Frag& a, const Frag& b, v8f c) {
    return __builtin_amdgcn_wmma_f32_16x16x32_bf16(
        false, a.bf, false, b.bf, (short)0, c, false, false);
}

// ---------------------------------------------------------------------------
// Zero-init scratch (c1, c2, h1_0, h2_0 laid out contiguously at ws start).
// ---------------------------------------------------------------------------
__global__ void zero_kernel(v4u* __restrict__ p, int n16) {
    int i = blockIdx.x * blockDim.x + threadIdx.x;
    int stride = gridDim.x * blockDim.x;
    v4u z = {0u, 0u, 0u, 0u};
    for (; i < n16; i += stride) p[i] = z;
}

// ---------------------------------------------------------------------------
// Pack [W ; U] -> Bt[n][k] bf16, n in [0,4096), k in [0,Ktot).
// ---------------------------------------------------------------------------
__global__ void pack_weights_kernel(const float* __restrict__ W, int KW,
                                    const float* __restrict__ U,
                                    int kPadX, int Ktot,
                                    __bf16* __restrict__ Bt) {
    int idx = blockIdx.x * blockDim.x + threadIdx.x;
    int total = NOUT * Ktot;
    if (idx >= total) return;
    int n = idx / Ktot;
    int k = idx - n * Ktot;
    float v;
    if (k < kPadX) v = (k < KW) ? W[(size_t)k * NOUT + n] : 0.0f;
    else           v = U[(size_t)(k - kPadX) * NOUT + n];
    Bt[idx] = (__bf16)v;
}

// ---------------------------------------------------------------------------
// Embedding gather + pad to 128, f32 -> bf16.  xemb layout: [B*T][128].
// ---------------------------------------------------------------------------
__global__ void embed_pack_kernel(const int* __restrict__ ids,
                                  const float* __restrict__ emb,
                                  __bf16* __restrict__ xemb) {
    int idx = blockIdx.x * blockDim.x + threadIdx.x;
    int total = BATCH * TSTEPS * EMBP;
    if (idx >= total) return;
    int k  = idx & (EMBP - 1);
    int bt = idx >> 7;
    float v = 0.0f;
    if (k < EMBD) v = emb[(size_t)ids[bt] * EMBD + k];
    xemb[idx] = (__bf16)v;
}

// ---------------------------------------------------------------------------
// Fused LSTM step:  z = [x_t | h_prev] @ [W;U]^T_packed + b ; gates -> c,h.
// Grid (BATCH/64, UNITS/32), block 256 (8 waves).
// ---------------------------------------------------------------------------
__global__ __launch_bounds__(256) void lstm_step_kernel(
    const __bf16* __restrict__ X,    // phase-0 A rows, ldX elements apart
    int ldX, int Kx,                 // Kx = padded input width (mult of 64)
    const __bf16* __restrict__ Hin,  // phase-1 A rows [B,1024]
    const __bf16* __restrict__ Bt,   // packed weights [4096][Ktot]
    int Ktot,
    const float* __restrict__ bias,  // [4096] gate order i,f,g,o
    float*  __restrict__ Cst,        // [B,1024] cell state (in/out)
    __bf16* __restrict__ Hout)       // [B,1024] new hidden (bf16)
{
    __shared__ __bf16 Bsh[2][BROWS][BSTRIDE];

    const int tid    = threadIdx.x;
    const int lane   = tid & 31;
    const int wave   = tid >> 5;
    const int lane16 = lane & 15;
    const int half   = lane >> 4;

    const int m_base = blockIdx.x * 64 + (wave & 3) * 16;
    const int u_blk  = blockIdx.y * 32;
    const int usub   = (wave >> 2) * 16;

    // ---- staging role: thread covers LDS row srow, 32-elem slab spart ----
    const int srow  = tid >> 1;          // 0..127  (= gate*32 + unit_local)
    const int spart = tid & 1;           // 0/1 -> K sub-slab of 32 elems
    const __bf16* srcRow =
        Bt + (size_t)((srow >> 5) * UNITS + u_blk + (srow & 31)) * Ktot
           + spart * 32;
    const unsigned ldsBase = (unsigned)(unsigned long long)&Bsh[0][0][0];
    const unsigned ldsRow0 =
        ldsBase + (unsigned)((srow * BSTRIDE + spart * 32) * sizeof(__bf16));
    const unsigned bufBytes = (unsigned)(BUF_ELEMS * sizeof(__bf16));

    v8f acc[4];
#pragma unroll
    for (int g = 0; g < 4; ++g)
#pragma unroll
        for (int r = 0; r < 8; ++r) acc[g][r] = 0.0f;

    const int aRow = m_base + lane16;
    const int k0   = half * 8;                       // K-half offset in a 32-chunk
    const __bf16* aX = X   + (size_t)aRow * ldX   + k0;
    const __bf16* aH = Hin + (size_t)aRow * UNITS + k0;

    const int nchunks  = (Kx + UNITS) / KC;
    const int nchunksX = Kx / KC;

    // ---- async DMA: stage one K-chunk of B into LDS buffer `sel`.
    // Global and LDS strides are both 16B here, and INST_OFFSET applies to
    // both sides, so one base pair + offset immediates covers 64B/thread.
    auto stage = [&](int kc, int sel) {
        const unsigned     ldsb = ldsRow0 + (sel ? bufBytes : 0u);
        unsigned long long ga   = (unsigned long long)(srcRow + kc);
        asm volatile(
            "global_load_async_to_lds_b128 %0, %1, off\n\t"
            "global_load_async_to_lds_b128 %0, %1, off offset:16\n\t"
            "global_load_async_to_lds_b128 %0, %1, off offset:32\n\t"
            "global_load_async_to_lds_b128 %0, %1, off offset:48"
            :: "v"(ldsb), "v"(ga) : "memory");
    };

    // B fragment from LDS (distinct register sets -> schedulable clause)
    auto ldb = [&](int sel, int g, int kofs) {
        Frag f;
        const __bf16* p = &Bsh[sel][g * 32 + usub + lane16][kofs];
        f.q[0] = *(const v4u*)(p);
        f.q[1] = *(const v4u*)(p + 16);
        return f;
    };

    // One K-chunk of math: hoist 2 A frags + 8 B frags, then 8 WMMAs.
    auto chunk_core = [&](int sel, const __bf16* ab) {
        Frag a0, a1;
        a0.q[0] = *(const v4u*)(ab);       a0.q[1] = *(const v4u*)(ab + 16);
        a1.q[0] = *(const v4u*)(ab + 32);  a1.q[1] = *(const v4u*)(ab + 48);

        Frag b0 = ldb(sel, 0, k0),      b1 = ldb(sel, 1, k0);
        Frag b2 = ldb(sel, 2, k0),      b3 = ldb(sel, 3, k0);
        Frag b4 = ldb(sel, 0, 32 + k0), b5 = ldb(sel, 1, 32 + k0);
        Frag b6 = ldb(sel, 2, 32 + k0), b7 = ldb(sel, 3, 32 + k0);

        acc[0] = wmma_bf16(a0, b0, acc[0]);
        acc[1] = wmma_bf16(a0, b1, acc[1]);
        acc[2] = wmma_bf16(a0, b2, acc[2]);
        acc[3] = wmma_bf16(a0, b3, acc[3]);
        acc[0] = wmma_bf16(a1, b4, acc[0]);
        acc[1] = wmma_bf16(a1, b5, acc[1]);
        acc[2] = wmma_bf16(a1, b6, acc[2]);
        acc[3] = wmma_bf16(a1, b7, acc[3]);
    };

    // prologue: stage chunk 0, wait DMA, barrier
    stage(0, 0);
    asm volatile("s_wait_asynccnt 0x0" ::: "memory");
    __syncthreads();

    // Phase-split chunk loops with UNCONDITIONAL next-chunk staging; the
    // final chunk is peeled below (no stage / no trailing barrier needed).
    int c = 0;
    for (; c < nchunksX; ++c) {                 // input (x_t) phase
        const int sel = c & 1;
        stage((c + 1) * KC, sel ^ 1);           // DMA overlaps WMMAs
        chunk_core(sel, aX + c * KC);
        asm volatile("s_wait_asynccnt 0x0" ::: "memory");
        __syncthreads();
    }
    for (; c < nchunks - 1; ++c) {              // recurrent (h_prev) phase
        const int sel = c & 1;
        stage((c + 1) * KC, sel ^ 1);
        chunk_core(sel, aH + (c * KC - Kx));
        asm volatile("s_wait_asynccnt 0x0" ::: "memory");
        __syncthreads();
    }
    chunk_core((nchunks - 1) & 1, aH + ((nchunks - 1) * KC - Kx));  // peeled

    // ---- Epilogue: gates + cell/hidden update ----
    // C/D layout: VGPR r -> M = r + 8*half (lane split), N = lane16.
    const int n  = u_blk + usub + lane16;
    const float bi  = bias[n];
    const float bf_ = bias[UNITS + n];
    const float bg  = bias[2 * UNITS + n];
    const float bo  = bias[3 * UNITS + n];

#pragma unroll
    for (int r = 0; r < 8; ++r) {
        const int m = m_base + half * 8 + r;
        const size_t idx = (size_t)m * UNITS + n;
        const float iv = fsig (acc[0][r] + bi);
        const float fv = fsig (acc[1][r] + bf_);
        const float gv = ftanh(acc[2][r] + bg);
        const float ov = fsig (acc[3][r] + bo);
        const float cn = fv * Cst[idx] + iv * gv;
        Cst[idx]  = cn;
        Hout[idx] = (__bf16)(ov * ftanh(cn));
    }
}

// ---------------------------------------------------------------------------
// Final dense + sigmoid:  out[b] = sigmoid( h2[b] . Wfc + bfc )
// ---------------------------------------------------------------------------
__global__ void fc_kernel(const __bf16* __restrict__ H,
                          const float* __restrict__ Wfc,
                          const float* __restrict__ bfc,
                          float* __restrict__ out) {
    int b = blockIdx.x * blockDim.x + threadIdx.x;
    if (b >= BATCH) return;
    const __bf16* h = H + (size_t)b * UNITS;
    float s = bfc[0];
#pragma unroll 4
    for (int k = 0; k < UNITS; ++k) s += (float)h[k] * Wfc[k];
    out[b] = 1.0f / (1.0f + __expf(-s));
}

// ---------------------------------------------------------------------------
// Host launcher
// ---------------------------------------------------------------------------
extern "C" void kernel_launch(void* const* d_in, const int* in_sizes, int n_in,
                              void* d_out, int out_size, void* d_ws, size_t ws_size,
                              hipStream_t stream) {
    const int*   ids = (const int*)  d_in[0];
    const float* emb = (const float*)d_in[1];
    const float* W1  = (const float*)d_in[2];
    const float* U1  = (const float*)d_in[3];
    const float* b1  = (const float*)d_in[4];
    const float* W2  = (const float*)d_in[5];
    const float* U2  = (const float*)d_in[6];
    const float* b2  = (const float*)d_in[7];
    const float* Wfc = (const float*)d_in[8];
    const float* bfc = (const float*)d_in[9];

    // Workspace layout (all segment sizes multiples of 256B):
    //  [c1 f32 2MB][c2 f32 2MB][h1_0 1MB][h2_0 1MB][h1_1 1MB][h2_1 1MB]
    //  [Bt1 9.4MB][Bt2 16.8MB][xemb 10.5MB]
    const size_t C_BYTES = (size_t)BATCH * UNITS * 4;
    const size_t H_BYTES = (size_t)BATCH * UNITS * 2;
    char* ws = (char*)d_ws;
    float*  c1   = (float*)(ws);
    float*  c2   = (float*)(ws + C_BYTES);
    __bf16* h1_0 = (__bf16*)(ws + 2 * C_BYTES);
    __bf16* h2_0 = (__bf16*)(ws + 2 * C_BYTES + H_BYTES);
    __bf16* h1_1 = (__bf16*)(ws + 2 * C_BYTES + 2 * H_BYTES);
    __bf16* h2_1 = (__bf16*)(ws + 2 * C_BYTES + 3 * H_BYTES);
    __bf16* Bt1  = (__bf16*)(ws + 2 * C_BYTES + 4 * H_BYTES);
    __bf16* Bt2  = Bt1 + (size_t)NOUT * KTOT1;
    __bf16* xemb = Bt2 + (size_t)NOUT * KTOT2;

    __bf16* h1buf[2] = {h1_0, h1_1};
    __bf16* h2buf[2] = {h2_0, h2_1};

    // 1) zero c1, c2, h1_0, h2_0 (contiguous 6 MB at ws start)
    {
        int n16 = (int)((2 * C_BYTES + 2 * H_BYTES) / 16);
        zero_kernel<<<1024, 256, 0, stream>>>((v4u*)ws, n16);
    }
    // 2) pack weights (bf16, transposed N-major, K padded/contiguous)
    {
        int tot1 = NOUT * KTOT1;
        pack_weights_kernel<<<(tot1 + 255) / 256, 256, 0, stream>>>(
            W1, EMBD, U1, EMBP, KTOT1, Bt1);
        int tot2 = NOUT * KTOT2;
        pack_weights_kernel<<<(tot2 + 255) / 256, 256, 0, stream>>>(
            W2, UNITS, U2, UNITS, KTOT2, Bt2);
    }
    // 3) embedding gather + pad
    {
        int tot = BATCH * TSTEPS * EMBP;
        embed_pack_kernel<<<(tot + 255) / 256, 256, 0, stream>>>(ids, emb, xemb);
    }

    // 4) recurrent chain: 80 timesteps x 2 layers, h ping-ponged per step
    dim3 grid(BATCH / 64, UNITS / 32);
    for (int t = 0; t < TSTEPS; ++t) {
        int in = t & 1, out = 1 - in;
        lstm_step_kernel<<<grid, 256, 0, stream>>>(
            xemb + (size_t)t * EMBP, TSTEPS * EMBP, EMBP,
            h1buf[in], Bt1, KTOT1, b1, c1, h1buf[out]);
        lstm_step_kernel<<<grid, 256, 0, stream>>>(
            h1buf[out], UNITS, UNITS,
            h2buf[in], Bt2, KTOT2, b2, c2, h2buf[out]);
    }

    // 5) final dense + sigmoid (h2 after t=79 lives in h2buf[0])
    fc_kernel<<<(BATCH + 255) / 256, 256, 0, stream>>>(
        h2buf[0], Wfc, bfc, (float*)d_out);
}
